// ApproachNet_regression_view_fps_53523882443482
// MI455X (gfx1250) — matrix operand order repaired and checked by reading, not compiled, not run on hardware
//
#include <hip/hip_runtime.h>
#include <hip/hip_bf16.h>
#include <math.h>

typedef __attribute__((ext_vector_type(16))) __bf16 v16bf;
typedef __attribute__((ext_vector_type(8)))  float  v8f;

#define BATCH 4
#define NPTS  20000
#define CDIM  256
#define SMP   1024
#define NVIEW 300

// d_out layout offsets (float elements), reference tuple order:
// objectness(4,2,20000) graspness(4,20000) gxyz(4,1024,3) inds(4,1024)
// gfeat(4,256,1024) fp2g(4,1024) vp(4,1024,3) topview(4,1024) rot(4,1024,3,3)
#define OUT_OBJ   0
#define OUT_GRASP 160000
#define OUT_GXYZ  240000
#define OUT_INDS  252288
#define OUT_GFEAT 256384
#define OUT_FP2G  1304960
#define OUT_VP    1309056
#define OUT_TOPV  1321344
#define OUT_ROT   1325440

// ---------------- WMMA fragment builders (ISA 7.12.2 layouts) ----------------
// A: 16x32 bf16 (MxK). lane m = lane&15, half = lane>>4.
// VGPR r<4 : K = 2r + 8*half ; r>=4 : K = 16 + 2(r-4) + 8*half  (pairs)
__device__ __forceinline__ v16bf load_a_frag(const float* __restrict__ W,
                                             int m0, int k0, int lane) {
  int m = lane & 15, half = lane >> 4;
  const float* row = W + (size_t)(m0 + m) * CDIM + k0;
  v16bf a;
#pragma unroll
  for (int r = 0; r < 8; ++r) {
    int kb = (r < 4) ? (2 * r + 8 * half) : (16 + 2 * (r - 4) + 8 * half);
    a[2 * r]     = (__bf16)row[kb];
    a[2 * r + 1] = (__bf16)row[kb + 1];
  }
  return a;
}

// B: 32x16 bf16 (KxN), source row-major [K][ldn]. lane n = lane&15.
// VGPR r<4 : K = 2r + 8*half ; r>=4 : K = 16 + 2(r-4) + 8*half  (pairs)
__device__ __forceinline__ v16bf load_b_frag(const float* __restrict__ F,
                                             int ldn, int n0, int k0, int lane) {
  int n = lane & 15, half = lane >> 4;
  v16bf b;
#pragma unroll
  for (int r = 0; r < 8; ++r) {
    int k = k0 + ((r < 4) ? (2 * r + 8 * half) : (16 + 2 * (r - 4) + 8 * half));
    b[2 * r]     = (__bf16)F[(size_t)k * ldn + n0 + n];
    b[2 * r + 1] = (__bf16)F[(size_t)(k + 1) * ldn + n0 + n];
  }
  return b;
}

// ---------------- BN folding prep: scale[c], shift[c] (bias folded in) ------
__global__ void bnprep_kernel(const float* gh_b1, const float* gh_g, const float* gh_be,
                              const float* gh_m, const float* gh_v,
                              const float* c1_b, const float* bn_g, const float* bn_b,
                              const float* bn_m, const float* bn_v,
                              float* s1, float* o1, float* s2, float* o2) {
  int c = threadIdx.x;
  float sc1 = gh_g[c] / sqrtf(gh_v[c] + 1e-5f);
  s1[c] = sc1;
  o1[c] = gh_b1[c] * sc1 + gh_be[c] - gh_m[c] * sc1;
  float sc2 = bn_g[c] / sqrtf(bn_v[c] + 1e-5f);
  s2[c] = sc2;
  o2[c] = c1_b[c] * sc2 + bn_b[c] - bn_m[c] * sc2;
}

// ---------------- Stage 1: fused conv1->bn->relu->conv2 (graspness head) ----
// One wave per 16-point tile; 16 row-tiles (256 channels) accumulated in regs.
__global__ __launch_bounds__(256)
void gemm1_kernel(const float* __restrict__ feat,   // [B][256][N]
                  const float* __restrict__ W1,     // [256][256]
                  const float* __restrict__ s1, const float* __restrict__ o1,
                  const float* __restrict__ W2,     // [3][256]
                  const float* __restrict__ b2,     // [3]
                  float* __restrict__ out_obj,      // [B][2][N]
                  float* __restrict__ out_grasp,    // [B][N]
                  int* __restrict__ gmask) {        // [B][N]
  const int tiles_per_b = NPTS / 16;  // 1250
  int wave = blockIdx.x * (blockDim.x >> 5) + (threadIdx.x >> 5);
  int b = wave / tiles_per_b;
  int tile = wave - b * tiles_per_b;
  if (b >= BATCH) return;
  int lane = threadIdx.x & 31;
  int n0 = tile * 16;
  const float* F = feat + (size_t)b * CDIM * NPTS;

  v8f acc[16];
#pragma unroll
  for (int rt = 0; rt < 16; ++rt)
#pragma unroll
    for (int j = 0; j < 8; ++j) acc[rt][j] = 0.f;

#pragma unroll 2
  for (int ks = 0; ks < 8; ++ks) {
    int k0 = ks * 32;
    v16bf bf = load_b_frag(F, NPTS, n0, k0, lane);
#pragma unroll
    for (int rt = 0; rt < 16; ++rt) {
      v16bf af = load_a_frag(W1, rt * 16, k0, lane);
      acc[rt] = __builtin_amdgcn_wmma_f32_16x16x32_bf16(
          false, af, false, bf, (short)0, acc[rt], false, false);
    }
  }

  // Epilogue: BN+ReLU then 3x256 second conv, per-point across lane halves.
  int half = lane >> 4, n = lane & 15;
  float p0 = 0.f, p1 = 0.f, p2 = 0.f;
#pragma unroll
  for (int rt = 0; rt < 16; ++rt) {
#pragma unroll
    for (int j = 0; j < 8; ++j) {
      int c = rt * 16 + half * 8 + j;
      float h = fmaxf(acc[rt][j] * s1[c] + o1[c], 0.f);
      p0 += W2[c] * h;
      p1 += W2[CDIM + c] * h;
      p2 += W2[2 * CDIM + c] * h;
    }
  }
  p0 += __shfl_xor(p0, 16);
  p1 += __shfl_xor(p1, 16);
  p2 += __shfl_xor(p2, 16);
  if (half == 0) {
    int nn = n0 + n;
    float ob0 = p0 + b2[0], ob1 = p1 + b2[1], g = p2 + b2[2];
    out_obj[(size_t)b * 2 * NPTS + nn] = ob0;
    out_obj[(size_t)b * 2 * NPTS + NPTS + nn] = ob1;
    out_grasp[(size_t)b * NPTS + nn] = g;
    gmask[b * NPTS + nn] = (g > 0.1f && ob1 > ob0) ? 1 : 0;
  }
}

// ---------------- Masked farthest-point sampling (sequential core) ----------
#define FPS_T 1024
#define PPT   20   // ceil(20000/1024)

__device__ __forceinline__ void amax2(float& v, int& i, float v2, int i2) {
  if (v2 > v || (v2 == v && i2 < i)) { v = v2; i = i2; }
}
__device__ __forceinline__ void wave_amax(float& v, int& i) {
#pragma unroll
  for (int off = 16; off; off >>= 1) {
    float ov = __shfl_xor(v, off);
    int   oi = __shfl_xor(i, off);
    amax2(v, i, ov, oi);
  }
}

__global__ __launch_bounds__(FPS_T)
void fps_kernel(const float* __restrict__ xyz,   // [B][N][3]
                const int* __restrict__ gmask,   // [B][N]
                int* __restrict__ inds,          // [B][S] ws
                float* __restrict__ inds_f) {    // [B][S] d_out
  __shared__ float svals[32];
  __shared__ int   sidx[32];
  __shared__ int   sbest;
  __shared__ float sbx, sby, sbz;

  int b = blockIdx.x;
  int tid = threadIdx.x;
  const float* X = xyz + (size_t)b * NPTS * 3;
  const int* M = gmask + b * NPTS;

  float px[PPT], py[PPT], pz[PPT], dist[PPT];
#pragma unroll
  for (int t = 0; t < PPT; ++t) {
    int i = tid + t * FPS_T;
    if (i < NPTS) {
      px[t] = X[i * 3]; py[t] = X[i * 3 + 1]; pz[t] = X[i * 3 + 2];
      dist[t] = M[i] ? 1e10f : -__builtin_inff();
    } else {
      px[t] = py[t] = pz[t] = 0.f;
      dist[t] = -__builtin_inff();
    }
  }

  // first = argmax(mask) (first true index, 0 if none)
  float bv = -__builtin_inff(); int bi = 0x7fffffff;
#pragma unroll
  for (int t = 0; t < PPT; ++t) {
    int i = tid + t * FPS_T;
    if (i < NPTS) amax2(bv, bi, M[i] ? 1.f : 0.f, i);
  }
  wave_amax(bv, bi);
  int w = tid >> 5;
  if ((tid & 31) == 0) { svals[w] = bv; sidx[w] = bi; }
  __syncthreads();
  if (w == 0) {
    float vv = svals[tid & 31]; int ii = sidx[tid & 31];
    wave_amax(vv, ii);
    if (tid == 0) sbest = ii;
  }
  __syncthreads();
  int best = sbest;
  if (tid == 0) { inds[b * SMP] = best; inds_f[b * SMP] = (float)best; }
  if (tid == (best & (FPS_T - 1))) {
    int slot = best >> 10;
    sbx = px[slot]; sby = py[slot]; sbz = pz[slot];
  }
  __syncthreads();
  float lx = sbx, ly = sby, lz = sbz;
  __syncthreads();

  for (int s = 1; s < SMP; ++s) {
    bv = -__builtin_inff(); bi = 0x7fffffff;
#pragma unroll
    for (int t = 0; t < PPT; ++t) {
      float dx = px[t] - lx, dy = py[t] - ly, dz = pz[t] - lz;
      float d = dx * dx + dy * dy + dz * dz;
      float nd = fminf(dist[t], d);
      dist[t] = nd;
      int i = tid + t * FPS_T;
      if (i < NPTS) amax2(bv, bi, nd, i);
    }
    wave_amax(bv, bi);
    if ((tid & 31) == 0) { svals[w] = bv; sidx[w] = bi; }
    __syncthreads();
    if (w == 0) {
      float vv = svals[tid & 31]; int ii = sidx[tid & 31];
      wave_amax(vv, ii);
      if (tid == 0) sbest = ii;
    }
    __syncthreads();
    best = sbest;
    if (tid == 0) { inds[b * SMP + s] = best; inds_f[b * SMP + s] = (float)best; }
    if (tid == (best & (FPS_T - 1))) {
      int slot = best >> 10;
      sbx = px[slot]; sby = py[slot]; sbz = pz[slot];
    }
    __syncthreads();
    lx = sbx; ly = sby; lz = sbz;
    __syncthreads();
  }
}

// ---------------- Gather xyz / features / graspness -------------------------
__global__ void gather_kernel(const float* __restrict__ xyz,
                              const float* __restrict__ feat,
                              const float* __restrict__ grasp,   // [B][N] (from d_out)
                              const int* __restrict__ inds,      // [B][S]
                              float* __restrict__ gfeat,         // [B][C][S]
                              float* __restrict__ gxyz,          // [B][S][3]
                              float* __restrict__ fp2g) {        // [B][S]
  int e = blockIdx.x * blockDim.x + threadIdx.x;
  if (e >= BATCH * CDIM * SMP) return;
  int b = e / (CDIM * SMP);
  int rem = e - b * (CDIM * SMP);
  int c = rem / SMP;
  int s = rem - c * SMP;
  int ind = inds[b * SMP + s];
  gfeat[e] = feat[((size_t)b * CDIM + c) * NPTS + ind];
  if (c < 3) gxyz[((size_t)b * SMP + s) * 3 + c] = xyz[((size_t)b * NPTS + ind) * 3 + c];
  if (c == 0) fp2g[b * SMP + s] = grasp[(size_t)b * NPTS + ind];
}

// ---------------- Stage 3: fused conv1->bn->relu->conv2 (view regression) ---
__global__ __launch_bounds__(256)
void gemm2_kernel(const float* __restrict__ gfeat,  // [B][256][1024]
                  const float* __restrict__ W1,     // c1_w [256][256]
                  const float* __restrict__ s2, const float* __restrict__ o2,
                  const float* __restrict__ W2,     // c2_w [3][256]
                  const float* __restrict__ b2,     // c2_b [3]
                  float* __restrict__ vp) {         // [B][S][3]
  const int tiles_per_b = SMP / 16;  // 64
  int wave = blockIdx.x * (blockDim.x >> 5) + (threadIdx.x >> 5);
  int b = wave / tiles_per_b;
  int tile = wave - b * tiles_per_b;
  if (b >= BATCH) return;
  int lane = threadIdx.x & 31;
  int n0 = tile * 16;
  const float* F = gfeat + (size_t)b * CDIM * SMP;

  v8f acc[16];
#pragma unroll
  for (int rt = 0; rt < 16; ++rt)
#pragma unroll
    for (int j = 0; j < 8; ++j) acc[rt][j] = 0.f;

#pragma unroll 2
  for (int ks = 0; ks < 8; ++ks) {
    int k0 = ks * 32;
    v16bf bf = load_b_frag(F, SMP, n0, k0, lane);
#pragma unroll
    for (int rt = 0; rt < 16; ++rt) {
      v16bf af = load_a_frag(W1, rt * 16, k0, lane);
      acc[rt] = __builtin_amdgcn_wmma_f32_16x16x32_bf16(
          false, af, false, bf, (short)0, acc[rt], false, false);
    }
  }

  int half = lane >> 4, n = lane & 15;
  float p0 = 0.f, p1 = 0.f, p2 = 0.f;
#pragma unroll
  for (int rt = 0; rt < 16; ++rt) {
#pragma unroll
    for (int j = 0; j < 8; ++j) {
      int c = rt * 16 + half * 8 + j;
      float h = fmaxf(acc[rt][j] * s2[c] + o2[c], 0.f);
      p0 += W2[c] * h;
      p1 += W2[CDIM + c] * h;
      p2 += W2[2 * CDIM + c] * h;
    }
  }
  p0 += __shfl_xor(p0, 16);
  p1 += __shfl_xor(p1, 16);
  p2 += __shfl_xor(p2, 16);
  if (half == 0) {
    int s = n0 + n;
    size_t off = ((size_t)b * SMP + s) * 3;
    vp[off]     = p0 + b2[0];
    vp[off + 1] = p1 + b2[1];
    vp[off + 2] = p2 + b2[2];
  }
}

// ---------------- Top view argmax (Fibonacci sphere regenerated) + rotation -
__global__ void view_kernel(const float* __restrict__ vp,
                            float* __restrict__ topv,
                            float* __restrict__ rot) {
  int e = blockIdx.x * blockDim.x + threadIdx.x;
  if (e >= BATCH * SMP) return;
  float x = vp[e * 3], y = vp[e * 3 + 1], z = vp[e * 3 + 2];
  float inv = 1.f / fmaxf(sqrtf(x * x + y * y + z * z), 1e-8f);
  float vx = x * inv, vy = y * inv, vz = z * inv;

  const float PHI = 0.6180339887498949f;    // (sqrt(5)-1)/2
  const float TWO_PI = 6.283185307179586f;
  float bestc = -__builtin_inff(); int besti = 0;
  for (int i = 0; i < NVIEW; ++i) {
    float tz = (2.f * (float)i + 1.f) / (float)NVIEW - 1.f;
    float r = sqrtf(fmaxf(1.f - tz * tz, 0.f));
    float ang = TWO_PI * (float)i * PHI;
    float tx = r * cosf(ang), ty = r * sinf(ang);
    float ti = 1.f / fmaxf(sqrtf(tx * tx + ty * ty + tz * tz), 1e-8f);
    float cs = (vx * tx + vy * ty + vz * tz) * ti;
    if (cs > bestc) { bestc = cs; besti = i; }
  }
  topv[e] = (float)besti;

  // rotation: towards = -vp
  float ax0 = -x, ax1 = -y, ax2 = -z;
  float ay0 = -ax1, ay1 = ax0, ay2 = 0.f;
  if (ay0 * ay0 + ay1 * ay1 == 0.f) { ay0 = 0.f; ay1 = 1.f; ay2 = 0.f; }
  float an = sqrtf(ax0 * ax0 + ax1 * ax1 + ax2 * ax2);
  float nx0 = ax0 / an, nx1 = ax1 / an, nx2 = ax2 / an;
  float yn = sqrtf(ay0 * ay0 + ay1 * ay1 + ay2 * ay2);
  float ny0 = ay0 / yn, ny1 = ay1 / yn, ny2 = ay2 / yn;
  float nz0 = nx1 * ny2 - nx2 * ny1;
  float nz1 = nx2 * ny0 - nx0 * ny2;
  float nz2 = nx0 * ny1 - nx1 * ny0;
  float* R = rot + (size_t)e * 9;   // stack([nx,ny,nz], axis=-1): columns
  R[0] = nx0; R[1] = ny0; R[2] = nz0;
  R[3] = nx1; R[4] = ny1; R[5] = nz1;
  R[6] = nx2; R[7] = ny2; R[8] = nz2;
}

// -----------------------------------------------------------------------------
extern "C" void kernel_launch(void* const* d_in, const int* in_sizes, int n_in,
                              void* d_out, int out_size, void* d_ws, size_t ws_size,
                              hipStream_t stream) {
  const float* seed_xyz  = (const float*)d_in[0];
  const float* seed_feat = (const float*)d_in[1];
  const float* gh_w1 = (const float*)d_in[2];
  const float* gh_b1 = (const float*)d_in[3];
  const float* gh_g  = (const float*)d_in[4];
  const float* gh_be = (const float*)d_in[5];
  const float* gh_m  = (const float*)d_in[6];
  const float* gh_v  = (const float*)d_in[7];
  const float* gh_w2 = (const float*)d_in[8];
  const float* gh_b2 = (const float*)d_in[9];
  const float* c1_w  = (const float*)d_in[10];
  const float* c1_b  = (const float*)d_in[11];
  const float* bn_g  = (const float*)d_in[12];
  const float* bn_b  = (const float*)d_in[13];
  const float* bn_m  = (const float*)d_in[14];
  const float* bn_v  = (const float*)d_in[15];
  const float* c2_w  = (const float*)d_in[16];
  const float* c2_b  = (const float*)d_in[17];

  float* out = (float*)d_out;
  float* out_obj   = out + OUT_OBJ;
  float* out_grasp = out + OUT_GRASP;
  float* out_gxyz  = out + OUT_GXYZ;
  float* out_inds  = out + OUT_INDS;
  float* out_gfeat = out + OUT_GFEAT;
  float* out_fp2g  = out + OUT_FP2G;
  float* out_vp    = out + OUT_VP;
  float* out_topv  = out + OUT_TOPV;
  float* out_rot   = out + OUT_ROT;

  float* wsf = (float*)d_ws;
  float* s1 = wsf;            // 256
  float* o1 = wsf + 256;      // 256
  float* s2 = wsf + 512;      // 256
  float* o2 = wsf + 768;      // 256
  int* gmask = (int*)(wsf + 1024);        // B*N = 80000
  int* inds  = gmask + BATCH * NPTS;      // B*S = 4096

  bnprep_kernel<<<1, 256, 0, stream>>>(gh_b1, gh_g, gh_be, gh_m, gh_v,
                                       c1_b, bn_g, bn_b, bn_m, bn_v,
                                       s1, o1, s2, o2);

  // 4 batches * 1250 point-tiles = 5000 waves / 8 waves per block
  gemm1_kernel<<<625, 256, 0, stream>>>(seed_feat, gh_w1, s1, o1, gh_w2, gh_b2,
                                        out_obj, out_grasp, gmask);

  fps_kernel<<<BATCH, FPS_T, 0, stream>>>(seed_xyz, gmask, inds, out_inds);

  int gather_n = BATCH * CDIM * SMP;
  gather_kernel<<<(gather_n + 255) / 256, 256, 0, stream>>>(
      seed_xyz, seed_feat, out_grasp, inds, out_gfeat, out_gxyz, out_fp2g);

  // 4 batches * 64 tiles = 256 waves / 8 per block
  gemm2_kernel<<<32, 256, 0, stream>>>(out_gfeat, c1_w, s2, o2, c2_w, c2_b, out_vp);

  view_kernel<<<(BATCH * SMP + 255) / 256, 256, 0, stream>>>(out_vp, out_topv, out_rot);
}
